// Block_head_39067022525006
// MI455X (gfx1250) — matrix operand order repaired and checked by reading, not compiled, not run on hardware
//
#include <hip/hip_runtime.h>
#include <hip/hip_bf16.h>
#include <math.h>

#define Bsz  32
#define Lsz  2048
#define Dsz  512
#define DFF  256
#define NBLK 6
#define EPSV 1e-6f
#define CHK  16               // L-chunks per batch
#define LCHK (Lsz / CHK)      // 128 keys per chunk
#define PARTS (CHK * 4)       // av_kernel partials per batch (4 lgroups/chunk)

typedef __attribute__((ext_vector_type(16))) __bf16 bf16x16;
typedef __attribute__((ext_vector_type(8)))  __bf16 bf16x8;
typedef __attribute__((ext_vector_type(8)))  float  f32x8;

__device__ __forceinline__ unsigned short f2bf(float f) {
    unsigned int u = __float_as_uint(f);
    u += 0x7FFFu + ((u >> 16) & 1u);          // round-to-nearest-even
    return (unsigned short)(u >> 16);
}
__device__ __forceinline__ float bf2f(unsigned short u) {
    return __uint_as_float(((unsigned int)u) << 16);
}

struct us4 { unsigned short x, y, z, w; };

// ---------------------------------------------------------------------------
// K,V f32 -> bf16 (one streaming pass; bf16 K+V = 128MB fits the 192MB L2)
// ---------------------------------------------------------------------------
__global__ void cvt_kv_kernel(const float* __restrict__ k, const float* __restrict__ v,
                              unsigned short* __restrict__ kb, unsigned short* __restrict__ vb,
                              size_t n4) {
    size_t i = (size_t)blockIdx.x * blockDim.x + threadIdx.x;
    size_t st = (size_t)gridDim.x * blockDim.x;
    for (size_t j = i; j < n4; j += st) {
        float4 kf = ((const float4*)k)[j];
        float4 vf = ((const float4*)v)[j];
        us4 ko = { f2bf(kf.x), f2bf(kf.y), f2bf(kf.z), f2bf(kf.w) };
        us4 vo = { f2bf(vf.x), f2bf(vf.y), f2bf(vf.z), f2bf(vf.w) };
        ((us4*)kb)[j] = ko;
        ((us4*)vb)[j] = vo;
    }
}

// ---------------------------------------------------------------------------
// W1 (blk,D,DFF) -> W1T (blk,DFF,D) bf16 ; W2 (blk,DFF,D) -> W2T (blk,D,DFF)
// ---------------------------------------------------------------------------
__global__ void cvt_w_kernel(const float* __restrict__ W1, const float* __restrict__ W2,
                             unsigned short* __restrict__ W1T, unsigned short* __restrict__ W2T) {
    int i = blockIdx.x * blockDim.x + threadIdx.x;
    if (i >= NBLK * DFF * Dsz) return;
    int blk = i / (DFF * Dsz);
    int r   = i % (DFF * Dsz);
    { // W1T[blk][n][k] = W1[blk][k][n]
        int n = r / Dsz, kk = r % Dsz;
        W1T[i] = f2bf(W1[(size_t)blk * Dsz * DFF + (size_t)kk * DFF + n]);
    }
    { // W2T[blk][n][k] = W2[blk][k][n]
        int n = r / DFF, kk = r % DFF;
        W2T[i] = f2bf(W2[(size_t)blk * DFF * Dsz + (size_t)kk * Dsz + n]);
    }
}

// ---------------------------------------------------------------------------
// 1) Raw scores: grid = B*CHK workgroups, 256 threads (8 waves), wave per key.
// ---------------------------------------------------------------------------
__global__ __launch_bounds__(256)
void scores_kernel(const float* __restrict__ qin, const unsigned short* __restrict__ Kb,
                   float* __restrict__ sc) {
    const int b    = blockIdx.x >> 4;
    const int c    = blockIdx.x & (CHK - 1);
    const int tid  = threadIdx.x;
    const int lane = tid & 31;
    const int wave = tid >> 5;

    const float* qb = qin + b * Dsz;
    float qreg[16];
#pragma unroll
    for (int j = 0; j < 16; ++j) qreg[j] = qb[lane * 16 + j];

    const unsigned short* Kbb = Kb + (size_t)b * Lsz * Dsz;
    const int l0 = c * LCHK;
    for (int l = l0 + wave; l < l0 + LCHK; l += 8) {
        const unsigned int* kr = (const unsigned int*)(Kbb + (size_t)l * Dsz + lane * 16);
        float acc = 0.f;
#pragma unroll
        for (int j = 0; j < 8; ++j) {
            unsigned int w = kr[j];
            acc = fmaf(qreg[2 * j],     bf2f((unsigned short)(w & 0xffffu)), acc);
            acc = fmaf(qreg[2 * j + 1], bf2f((unsigned short)(w >> 16)),     acc);
        }
        for (int o = 16; o > 0; o >>= 1) acc += __shfl_xor(acc, o, 32);
        if (lane == 0) sc[(size_t)b * Lsz + l] = acc * 0.04419417382415922f; // 1/sqrt(512)
    }
}

// ---------------------------------------------------------------------------
// 2) Softmax over L (in place) + write weights output. grid = B, 256 thr.
// ---------------------------------------------------------------------------
__global__ __launch_bounds__(256)
void softmax_kernel(float* __restrict__ sc, float* __restrict__ wout, int idx) {
    __shared__ float red[8];
    const int b = blockIdx.x, tid = threadIdx.x, lane = tid & 31, wave = tid >> 5;
    float* sb = sc + (size_t)b * Lsz;

    float mx = -INFINITY;
    for (int l = tid; l < Lsz; l += 256) mx = fmaxf(mx, sb[l]);
    for (int o = 16; o > 0; o >>= 1) mx = fmaxf(mx, __shfl_xor(mx, o, 32));
    if (lane == 0) red[wave] = mx;
    __syncthreads();
#pragma unroll
    for (int i = 0; i < 8; ++i) mx = fmaxf(mx, red[i]);
    __syncthreads();

    float s = 0.f;
    for (int l = tid; l < Lsz; l += 256) { float e = __expf(sb[l] - mx); sb[l] = e; s += e; }
    for (int o = 16; o > 0; o >>= 1) s += __shfl_xor(s, o, 32);
    if (lane == 0) red[wave] = s;
    __syncthreads();
    float tot = 0.f;
#pragma unroll
    for (int i = 0; i < 8; ++i) tot += red[i];
    float inv = 1.f / tot;

    for (int l = tid; l < Lsz; l += 256) {
        float p = sb[l] * inv;
        sb[l] = p;
        wout[((size_t)b * Lsz + l) * 6 + idx] = p;
    }
}

// ---------------------------------------------------------------------------
// 3) Partial A = p . V over one L-chunk. grid = B*CHK, 256 thr.
//    Thread layout: lgroup = t>>6 (4-way split of the chunk's rows),
//    8 contiguous d's per thread -> one b128 load per row per thread.
// ---------------------------------------------------------------------------
__global__ __launch_bounds__(256)
void av_kernel(const float* __restrict__ sc, const unsigned short* __restrict__ Vb,
               float* __restrict__ Apart) {
    __shared__ float p[LCHK];
    const int b  = blockIdx.x >> 4;
    const int c  = blockIdx.x & (CHK - 1);
    const int t  = threadIdx.x;
    const int l0 = c * LCHK;

    if (t < LCHK) p[t] = sc[(size_t)b * Lsz + l0 + t];
    __syncthreads();

    const unsigned short* Vbb = Vb + ((size_t)b * Lsz + l0) * Dsz;
    const int lg = t >> 6;          // 0..3
    const int d0 = (t & 63) * 8;    // 8 contiguous d's (16 bytes)

    float acc[8];
#pragma unroll
    for (int j = 0; j < 8; ++j) acc[j] = 0.f;

    for (int l = lg; l < LCHK; l += 4) {
        if ((l & 31) == lg)
            __builtin_prefetch(Vbb + (size_t)(l + 32) * Dsz + d0, 0, 3);
        uint4 w = *(const uint4*)(Vbb + (size_t)l * Dsz + d0);
        float pl = p[l];
        acc[0] = fmaf(pl, bf2f((unsigned short)(w.x & 0xffffu)), acc[0]);
        acc[1] = fmaf(pl, bf2f((unsigned short)(w.x >> 16)),     acc[1]);
        acc[2] = fmaf(pl, bf2f((unsigned short)(w.y & 0xffffu)), acc[2]);
        acc[3] = fmaf(pl, bf2f((unsigned short)(w.y >> 16)),     acc[3]);
        acc[4] = fmaf(pl, bf2f((unsigned short)(w.z & 0xffffu)), acc[4]);
        acc[5] = fmaf(pl, bf2f((unsigned short)(w.z >> 16)),     acc[5]);
        acc[6] = fmaf(pl, bf2f((unsigned short)(w.w & 0xffffu)), acc[6]);
        acc[7] = fmaf(pl, bf2f((unsigned short)(w.w >> 16)),     acc[7]);
    }

    float* Ap = Apart + ((size_t)b * PARTS + c * 4 + lg) * Dsz + d0;
    float4 o0 = { acc[0], acc[1], acc[2], acc[3] };
    float4 o1 = { acc[4], acc[5], acc[6], acc[7] };
    ((float4*)Ap)[0] = o0;
    ((float4*)Ap)[1] = o1;
}

// ---------------------------------------------------------------------------
// 4) Reduce partials + residual + norm1. grid = B, 512 thr.
// ---------------------------------------------------------------------------
__global__ __launch_bounds__(512)
void norm1_kernel(const float* __restrict__ qin, const float* __restrict__ Apart,
                  const float* __restrict__ alpha1, const float* __restrict__ bias1,
                  float* __restrict__ qn1, unsigned short* __restrict__ qn1b) {
    __shared__ float s_m[16], s_v[16];
    const int b = blockIdx.x, tid = threadIdx.x, lane = tid & 31, wave = tid >> 5;

    float x = qin[b * Dsz + tid];
    const float* Ap = Apart + (size_t)b * PARTS * Dsz;
#pragma unroll 8
    for (int c = 0; c < PARTS; ++c) x += Ap[c * Dsz + tid];

    float sx = x, sxx = x * x;
    for (int o = 16; o > 0; o >>= 1) { sx += __shfl_xor(sx, o, 32); sxx += __shfl_xor(sxx, o, 32); }
    if (lane == 0) { s_m[wave] = sx; s_v[wave] = sxx; }
    __syncthreads();
    float tm = 0.f, tv = 0.f;
#pragma unroll
    for (int i = 0; i < 16; ++i) { tm += s_m[i]; tv += s_v[i]; }
    float mu  = tm / Dsz;
    float var = (tv - Dsz * mu * mu) / (Dsz - 1);
    float in2 = 1.f / (sqrtf(var) + EPSV);
    float y   = alpha1[tid] * (x - mu) * in2 + bias1[tid];
    qn1[b * Dsz + tid]  = y;
    qn1b[b * Dsz + tid] = f2bf(y);
}

// ---------------------------------------------------------------------------
// 5) FFN + norm2, WMMA bf16. One workgroup of 32 waves.
// A-frag (16-bit A 16x32): lane<16 -> row lane, K {0..7}U{16..23};
//                          lane>=16 -> row lane-16, K {8..15}U{24..31}.
// B-frag (dense 32x16):    lane<16 -> col lane, K {0..7}U{16..23}; lane>=16 -> K {8..15}U{24..31}.
// C tile: VGPR j, lane<16 -> (M=j, N=lane); lane>=16 -> (M=j+8, N=lane-16).
// ---------------------------------------------------------------------------
__global__ __launch_bounds__(1024)
void ffn_kernel(const float* __restrict__ qn1, const unsigned short* __restrict__ qn1b,
                const unsigned short* __restrict__ W1T, const unsigned short* __restrict__ W2T,
                const float* __restrict__ b1, const float* __restrict__ b2,
                const float* __restrict__ alpha2, const float* __restrict__ bias2,
                float* __restrict__ qout, float* __restrict__ out, int head) {
    __shared__ unsigned short s_h[Bsz][DFF + 8];
    __shared__ float s_f[Bsz][Dsz];

    const int tid  = threadIdx.x;
    const int lane = tid & 31;
    const int wave = tid >> 5;
    const int half = lane >> 4;
    const int l15  = lane & 15;

    { // GEMM1: one 16x16 tile per wave (2 Mt x 16 Nt)
        int mt = wave >> 4, nt = wave & 15;
        f32x8 c = {};
        int arow = mt * 16 + l15;
        int akb  = half ? 8 : 0;
        int bkb  = half ? 8 : 0;
        const unsigned short* Abase = qn1b + arow * Dsz;
        const unsigned short* Bbase = W1T + (size_t)(nt * 16 + l15) * Dsz;
        for (int k0 = 0; k0 < Dsz; k0 += 32) {
            bf16x8 a0 = *(const bf16x8*)(Abase + k0 + akb);
            bf16x8 a1 = *(const bf16x8*)(Abase + k0 + akb + 16);
            bf16x16 a = __builtin_shufflevector(a0, a1, 0,1,2,3,4,5,6,7,8,9,10,11,12,13,14,15);
            bf16x8 b0 = *(const bf16x8*)(Bbase + k0 + bkb);
            bf16x8 b1v = *(const bf16x8*)(Bbase + k0 + bkb + 16);
            bf16x16 bm = __builtin_shufflevector(b0, b1v, 0,1,2,3,4,5,6,7,8,9,10,11,12,13,14,15);
            c = __builtin_amdgcn_wmma_f32_16x16x32_bf16(false, a, false, bm,
                                                        (short)0, c, false, false);
        }
#pragma unroll
        for (int j = 0; j < 8; ++j) {
            int m = mt * 16 + j + half * 8;
            int n = nt * 16 + l15;
            s_h[m][n] = f2bf(fmaxf(c[j] + b1[n], 0.f));
        }
    }
    __syncthreads();

    // GEMM2: 64 tiles (2 Mt x 32 Nt), 2 per wave
    for (int t = wave; t < 64; t += 32) {
        int mt = t >> 5, nt = t & 31;
        f32x8 c = {};
        int arow = mt * 16 + l15;
        int akb  = half ? 8 : 0;
        int bkb  = half ? 8 : 0;
        const unsigned short* Bbase = W2T + (size_t)(nt * 16 + l15) * DFF;
        for (int k0 = 0; k0 < DFF; k0 += 32) {
            const unsigned short* Ap = &s_h[arow][k0 + akb];
            bf16x8 a0 = *(const bf16x8*)Ap;
            bf16x8 a1 = *(const bf16x8*)(Ap + 16);
            bf16x16 a = __builtin_shufflevector(a0, a1, 0,1,2,3,4,5,6,7,8,9,10,11,12,13,14,15);
            bf16x8 b0 = *(const bf16x8*)(Bbase + k0 + bkb);
            bf16x8 b1v = *(const bf16x8*)(Bbase + k0 + bkb + 16);
            bf16x16 bm = __builtin_shufflevector(b0, b1v, 0,1,2,3,4,5,6,7,8,9,10,11,12,13,14,15);
            c = __builtin_amdgcn_wmma_f32_16x16x32_bf16(false, a, false, bm,
                                                        (short)0, c, false, false);
        }
#pragma unroll
        for (int j = 0; j < 8; ++j) {
            int m = mt * 16 + j + half * 8;
            int n = nt * 16 + l15;
            s_f[m][n] = c[j] + b2[n] + qn1[m * Dsz + n];
        }
    }
    __syncthreads();

    { // norm2: wave w handles row m=w
        int m = wave;
        float xv[16];
        float sx = 0.f, sxx = 0.f;
#pragma unroll
        for (int j = 0; j < 16; ++j) {
            float v = s_f[m][lane + 32 * j];
            xv[j] = v; sx += v; sxx += v * v;
        }
        for (int o = 16; o > 0; o >>= 1) { sx += __shfl_xor(sx, o, 32); sxx += __shfl_xor(sxx, o, 32); }
        float mu  = sx / Dsz;
        float var = (sxx - Dsz * mu * mu) / (Dsz - 1);
        float in2 = 1.f / (sqrtf(var) + EPSV);
#pragma unroll
        for (int j = 0; j < 16; ++j) {
            int n = lane + 32 * j;
            float y = alpha2[n] * (xv[j] - mu) * in2 + bias2[n];
            qout[m * Dsz + n] = y;
            if (head >= 0) out[m * (2 * Dsz) + head * Dsz + n] = y;
        }
    }
}

// ---------------------------------------------------------------------------
extern "C" void kernel_launch(void* const* d_in, const int* in_sizes, int n_in,
                              void* d_out, int out_size, void* d_ws, size_t ws_size,
                              hipStream_t stream) {
    const float* q   = (const float*)d_in[0];
    const float* k   = (const float*)d_in[1];
    const float* v   = (const float*)d_in[2];
    const float* W1  = (const float*)d_in[3];
    const float* b1  = (const float*)d_in[4];
    const float* W2  = (const float*)d_in[5];
    const float* b2  = (const float*)d_in[6];
    const float* a1  = (const float*)d_in[7];
    const float* bi1 = (const float*)d_in[8];
    const float* a2  = (const float*)d_in[9];
    const float* bi2 = (const float*)d_in[10];

    // workspace layout (bytes)
    const size_t KV = (size_t)Bsz * Lsz * Dsz * 2;        // 64 MB each
    const size_t WT = (size_t)NBLK * DFF * Dsz * 2;       // 1.5 MB each
    char* ws = (char*)d_ws;
    unsigned short* Kb    = (unsigned short*)(ws);
    unsigned short* Vb    = (unsigned short*)(ws + KV);
    unsigned short* W1T   = (unsigned short*)(ws + 2 * KV);
    unsigned short* W2T   = (unsigned short*)(ws + 2 * KV + WT);
    float*          qn1   = (float*)(ws + 2 * KV + 2 * WT);
    unsigned short* qn1b  = (unsigned short*)((char*)qn1 + (size_t)Bsz * Dsz * 4);
    float*          qcur  = (float*)((char*)qn1b + (size_t)Bsz * Dsz * 2);
    float*          sc    = (float*)((char*)qcur + (size_t)Bsz * Dsz * 4);          // B*L f32
    float*          Apart = (float*)((char*)sc + (size_t)Bsz * Lsz * 4);            // B*PARTS*D f32

    float* out  = (float*)d_out;                 // (B, 2D)
    float* wout = out + Bsz * 2 * Dsz;           // (B, L, 6)

    size_t n4 = (size_t)Bsz * Lsz * Dsz / 4;
    cvt_kv_kernel<<<8192, 256, 0, stream>>>(k, v, Kb, Vb, n4);
    cvt_w_kernel<<<(NBLK * DFF * Dsz + 255) / 256, 256, 0, stream>>>(W1, W2, W1T, W2T);

    for (int idx = 0; idx < NBLK; ++idx) {
        const float* qsrc = (idx == 0) ? q : qcur;
        scores_kernel<<<Bsz * CHK, 256, 0, stream>>>(qsrc, Kb, sc);
        softmax_kernel<<<Bsz, 256, 0, stream>>>(sc, wout, idx);
        av_kernel<<<Bsz * CHK, 256, 0, stream>>>(sc, Vb, Apart);
        norm1_kernel<<<Bsz, 512, 0, stream>>>(qsrc, Apart, a1 + idx * Dsz, bi1 + idx * Dsz,
                                              qn1, qn1b);
        int head = (idx == 2) ? 0 : (idx == 5) ? 1 : -1;
        ffn_kernel<<<1, 1024, 0, stream>>>(qn1, qn1b,
                                           W1T + (size_t)idx * DFF * Dsz,
                                           W2T + (size_t)idx * Dsz * DFF,
                                           b1 + idx * DFF, b2 + idx * Dsz,
                                           a2 + idx * Dsz, bi2 + idx * Dsz,
                                           qcur, out, head);
    }
}
